// LSTM_VAR_29729763623486
// MI455X (gfx1250) — matrix operand order repaired and checked by reading, not compiled, not run on hardware
//
#include <hip/hip_runtime.h>
#include <hip/hip_bf16.h>

typedef __attribute__((ext_vector_type(16))) _Float16 v16h;
typedef __attribute__((ext_vector_type(8)))  _Float16 v8h;
typedef __attribute__((ext_vector_type(8)))  float    v8f;

#define B_TOT   512
#define T_LEN   256
#define D_IN    64
#define UNITSC  256
#define BT      32            // batch rows per group
#define GROUPW  8             // workgroups per group (each owns 32 cells)
#define NGROUP  (B_TOT / BT)  // 16 groups -> 128 WGs
#define XS      72            // xbuf LDS stride (halves)
#define HS      264           // h LDS stride (halves)
#define ZS      132           // zbuf LDS stride (floats)
#define NTT     64            // 1024/16 global N-tiles

#define KT0     10            // (64+256)/32 K-tiles, layer 0
#define KT1     16            // (256+256)/32 K-tiles, layer 1
#define WP0_HALVES (KT0 * NTT * 32 * 16)     // 327680
#define WP1_HALVES (KT1 * NTT * 32 * 16)     // 524288
#define WSL0_HALVES (KT0 * 8 * 32 * 16)      // 40960 (80 KB) per-WG slice
#define WSL1_HALVES (KT1 * 8 * 32 * 16)      // 65536 (128 KB) per-WG slice

// workspace layout (bytes)
#define WP0_OFF 0
#define WP1_OFF (WP0_HALVES * 2)                       // 655360
#define HG0_OFF (WP1_OFF + WP1_HALVES * 2)             // h0 broadcast, 2x16x32x256 f16
#define HG1_OFF (HG0_OFF + 2 * NGROUP * BT * UNITSC * 2)
#define BAR_OFF (HG1_OFF + 2 * NGROUP * BT * UNITSC * 2)
#define BAR_BYTES (NGROUP * 512 * 4)

// dynamic LDS partition (bytes): wsl0|wsl1|hbuf0|hbuf1|xbuf|zbuf
#define SMEM_BYTES ((WSL0_HALVES + WSL1_HALVES + 2 * BT * HS + BT * XS) * 2 + BT * ZS * 4)

// ---------------------------------------------------------------------------
// CDNA5 async global->LDS copy (16B per lane, ASYNCcnt-tracked).
// LDS destination address = low 32 bits of the generic shared pointer
// (ISA 10.2: generic->LDS truncates to addr[31:0]).
// ---------------------------------------------------------------------------
__device__ __forceinline__ void async_copy16(void* ldsDst, const void* gSrc)
{
    unsigned l = (unsigned)(uintptr_t)ldsDst;
    asm volatile("global_load_async_to_lds_b128 %0, %1, off"
                 :: "v"(l), "v"(gSrc) : "memory");
}
__device__ __forceinline__ void wait_async0()
{
    asm volatile("s_wait_asynccnt 0x0" ::: "memory");
}

// ---------------------------------------------------------------------------
// Kernel 1: pack [W;U] (f32 KxN row-major) into f16 WMMA-B fragment tiles.
// Tile (kt,ntIdx): 32(K)x16(N); lane L holds 16 consecutive K of column
// ntIdx*16+(L&15), K base kt*32+(L>>4)*16 -> one contiguous 32B chunk/lane.
// ---------------------------------------------------------------------------
__global__ void pack_weights(const float* __restrict__ W0, const float* __restrict__ U0,
                             const float* __restrict__ W1, const float* __restrict__ U1,
                             _Float16* __restrict__ wp0, _Float16* __restrict__ wp1)
{
    int idx = blockIdx.x * 256 + threadIdx.x;
    if (idx < WP0_HALVES) {
        int j = idx & 15, lane = (idx >> 4) & 31, tile = idx >> 9;
        int kt = tile >> 6, ntIdx = tile & 63;
        int k = kt * 32 + ((lane >> 4) << 4) + j;
        int n = (ntIdx << 4) + (lane & 15);
        float v = (k < D_IN) ? W0[(size_t)k * 1024 + n]
                             : U0[(size_t)(k - D_IN) * 1024 + n];
        wp0[idx] = (_Float16)v;
    } else if (idx < WP0_HALVES + WP1_HALVES) {
        int i = idx - WP0_HALVES;
        int j = i & 15, lane = (i >> 4) & 31, tile = i >> 9;
        int kt = tile >> 6, ntIdx = tile & 63;
        int k = kt * 32 + ((lane >> 4) << 4) + j;
        int n = (ntIdx << 4) + (lane & 15);
        float v = (k < UNITSC) ? W1[(size_t)k * 1024 + n]
                               : U1[(size_t)(k - UNITSC) * 1024 + n];
        wp1[i] = (_Float16)v;
    }
}

// ---------------------------------------------------------------------------
// z-slice GEMM: z(32 x 128) = [AsegA | AsegB] @ Wslice, weights LDS-resident.
// Wave w computes one 16-wide N-tile (gate w>>1, cell-half w&1) -> zbuf.
// ---------------------------------------------------------------------------
template <int KTA, int KTB>
__device__ __forceinline__ void layer_gemm(const _Float16* asegA, int strA,
                                           const _Float16* asegB, int strB,
                                           const _Float16* wsl, float* zbuf,
                                           int lane, int wave)
{
    v8f acc[2] = {};
    const int mrow = lane & 15;
    const int sel  = lane >> 4;

#pragma unroll
    for (int kt = 0; kt < KTA + KTB; ++kt) {
        const _Float16* as;
        int astr, k0;
        if (kt < KTA) { as = asegA; astr = strA; k0 = kt * 32; }
        else          { as = asegB; astr = strB; k0 = (kt - KTA) * 32; }

        v16h af[2];
#pragma unroll
        for (int mt = 0; mt < 2; ++mt) {
            const _Float16* p = as + (mt * 16 + mrow) * astr + k0 + sel * 16;
            union { v16h v; v8h h[2]; } u;
            u.h[0] = *(const v8h*)(p);
            u.h[1] = *(const v8h*)(p + 8);
            af[mt] = u.v;
        }
        v16h bf = *(const v16h*)(wsl + ((size_t)(kt * 8 + wave) * 32 + lane) * 16);
        acc[0] = __builtin_amdgcn_wmma_f32_16x16x32_f16(false, af[0], false, bf,
                                                        (short)0, acc[0], false, false);
        acc[1] = __builtin_amdgcn_wmma_f32_16x16x32_f16(false, af[1], false, bf,
                                                        (short)0, acc[1], false, false);
    }

    const int ncol = lane & 15;
#pragma unroll
    for (int mt = 0; mt < 2; ++mt)
#pragma unroll
        for (int e = 0; e < 8; ++e) {
            int row = mt * 16 + e + (sel << 3);      // C layout: M = e (+8 hi half)
            zbuf[row * ZS + wave * 16 + ncol] = acc[mt][e];
        }
}

// ---------------------------------------------------------------------------
// Gate fusion: each thread owns 4 (row, cell) elements (cell fixed = tid&31).
// Reads i/f/g/o from zbuf, updates c (VGPRs), stores h slice (f16) to hglob.
// ---------------------------------------------------------------------------
__device__ __forceinline__ void gate_fuse(const float* zbuf, float bi, float bf,
                                          float bg, float bo, float cst[4],
                                          _Float16* hg, int hgRowBase, int ucell,
                                          int tid)
{
    const int cl = tid & 31;
#pragma unroll
    for (int q = 0; q < 4; ++q) {
        int row = (tid >> 5) + q * 8;
        float zi = zbuf[row * ZS +       cl] + bi;
        float zf = zbuf[row * ZS +  32 + cl] + bf;
        float zg = zbuf[row * ZS +  64 + cl] + bg;
        float zo = zbuf[row * ZS +  96 + cl] + bo;
        float ig = 1.f / (1.f + __expf(-zi));
        float fg = 1.f / (1.f + __expf(-zf));
        float gg = tanhf(zg);
        float og = 1.f / (1.f + __expf(-zo));
        float c  = fg * cst[q] + ig * gg;
        cst[q]   = c;
        float h  = og * tanhf(c);
        hg[(size_t)(hgRowBase + row) * UNITSC + ucell] = (_Float16)h;
    }
}

// ---------------------------------------------------------------------------
// Kernel 2: fused 2-layer LSTM. 16 groups x 8 WGs; group owns 32 batch rows,
// WG j owns 32 cells (z-cols g*256+32j..+32 for all gates). Weight slices are
// LDS-resident (208 KB); per step the 16 KB h vector is exchanged through L2
// with a monotonic atomic group barrier. Layer-1 barrier is split arrive/wait
// so its latency hides under next step's layer-0 compute. All global->LDS
// transfers use the CDNA5 async-to-LDS path (ASYNCcnt).
// ---------------------------------------------------------------------------
__global__ void __launch_bounds__(256) lstm_fused(const float* __restrict__ x,
                                                  const float* __restrict__ sscale,
                                                  const float* __restrict__ smin,
                                                  const float* __restrict__ b0,
                                                  const float* __restrict__ b1,
                                                  const _Float16* __restrict__ wp0,
                                                  const _Float16* __restrict__ wp1,
                                                  _Float16* __restrict__ hg0,
                                                  _Float16* __restrict__ hg1,
                                                  unsigned* __restrict__ bar)
{
    extern __shared__ char smem[];
    _Float16* wsl0  = (_Float16*)smem;
    _Float16* wsl1  = wsl0 + WSL0_HALVES;
    _Float16* hbuf0 = wsl1 + WSL1_HALVES;
    _Float16* hbuf1 = hbuf0 + BT * HS;
    _Float16* xbuf  = hbuf1 + BT * HS;
    float*    zbuf  = (float*)(xbuf + BT * XS);

    const int tid   = threadIdx.x;
    const int lane  = tid & 31;
    const int wave  = tid >> 5;
    const int group = blockIdx.x >> 3;
    const int j     = blockIdx.x & 7;
    const int r0    = group * BT;
    unsigned* gbar  = bar + group * 512;

    // ---- async-load this WG's weight slices into LDS (once), 16B units ----
    for (int c = tid; c < WSL0_HALVES / 8; c += 256) {
        int u = c & 63, tl = c >> 6, kt = tl >> 3, l = tl & 7;
        int ln = u >> 1, part = u & 1;
        int ntg = (l >> 1) * 16 + 2 * j + (l & 1);
        async_copy16(wsl0 + (size_t)c * 8,
                     wp0 + ((size_t)(kt * NTT + ntg) * 32 + ln) * 16 + part * 8);
    }
    for (int c = tid; c < WSL1_HALVES / 8; c += 256) {
        int u = c & 63, tl = c >> 6, kt = tl >> 3, l = tl & 7;
        int ln = u >> 1, part = u & 1;
        int ntg = (l >> 1) * 16 + 2 * j + (l & 1);
        async_copy16(wsl1 + (size_t)c * 8,
                     wp1 + ((size_t)(kt * NTT + ntg) * 32 + ln) * 16 + part * 8);
    }
    for (int i = tid; i < BT * HS; i += 256) { hbuf0[i] = (_Float16)0.f; hbuf1[i] = (_Float16)0.f; }

    float cst0[4] = {0.f, 0.f, 0.f, 0.f};
    float cst1[4] = {0.f, 0.f, 0.f, 0.f};

    const int cl    = tid & 31;
    const int ucell = 32 * j + cl;
    const float b0i = b0[          ucell], b0f = b0[256 + ucell];
    const float b0g = b0[512 + ucell],     b0o = b0[768 + ucell];
    const float b1i = b1[          ucell], b1f = b1[256 + ucell];
    const float b1g = b1[512 + ucell],     b1o = b1[768 + ucell];

    const int   dcol = tid & 63;
    const float sc   = sscale[dcol];
    const float mn   = smin[dcol];

    wait_async0();
    __syncthreads();

    for (int t = 0; t < T_LEN; ++t) {
        const int p = t & 1;
        // stage scaled x_t (32 x 64) as f16
#pragma unroll
        for (int i = tid; i < BT * D_IN; i += 256) {
            int row = i >> 6;
            float v = x[((size_t)(r0 + row) * T_LEN + t) * D_IN + dcol] * sc + mn;
            xbuf[row * XS + dcol] = (_Float16)v;
        }
        __syncthreads();

        // ---- layer 0: z = [x_t, h0] @ Wslice ----
        layer_gemm<2, 8>(xbuf, XS, hbuf0, HS, wsl0, zbuf, lane, wave);
        __syncthreads();
        gate_fuse(zbuf, b0i, b0f, b0g, b0o, cst0,
                  hg0, (p * NGROUP + group) * BT, ucell, tid);
        __threadfence();
        __syncthreads();
        if (tid == 0) {                                  // exposed h0 barrier
            atomicAdd(&gbar[t * 2], 1u);
            while (((volatile unsigned*)gbar)[t * 2] < GROUPW)
                __builtin_amdgcn_s_sleep(2);
        }
        __syncthreads();
        __threadfence();
        // async broadcast-read full h0 into LDS
        for (int c = tid; c < BT * UNITSC / 8; c += 256) {
            int row = c >> 5, o8 = (c & 31) * 8;
            async_copy16(hbuf0 + row * HS + o8,
                         hg0 + (size_t)((p * NGROUP + group) * BT + row) * UNITSC + o8);
        }
        // hidden h1 barrier: wait for step t-1 arrivals, then refresh hbuf1
        if (t > 0) {
            if (tid == 0) {
                while (((volatile unsigned*)gbar)[(t - 1) * 2 + 1] < GROUPW)
                    __builtin_amdgcn_s_sleep(2);
            }
            __syncthreads();
            __threadfence();
            const int pp = (t - 1) & 1;
            for (int c = tid; c < BT * UNITSC / 8; c += 256) {
                int row = c >> 5, o8 = (c & 31) * 8;
                async_copy16(hbuf1 + row * HS + o8,
                             hg1 + (size_t)((pp * NGROUP + group) * BT + row) * UNITSC + o8);
            }
        }
        wait_async0();
        __syncthreads();

        // ---- layer 1: z = [h0, h1] @ Wslice ----
        layer_gemm<8, 8>(hbuf0, HS, hbuf1, HS, wsl1, zbuf, lane, wave);
        __syncthreads();
        gate_fuse(zbuf, b1i, b1f, b1g, b1o, cst1,
                  hg1, (p * NGROUP + group) * BT, ucell, tid);
        __threadfence();
        __syncthreads();
        if (tid == 0) atomicAdd(&gbar[t * 2 + 1], 1u);   // arrive only; wait is deferred
        __syncthreads();
    }
}

// ---------------------------------------------------------------------------
// Kernel 3: heads (~0.7 GFLOP), one WG per batch row (kernel boundary is the
// global sync that makes all h1 slices visible).
// out = (x_flat[:, :2048] - h@F_nlb) @ F_l + h @ F_nl
// ---------------------------------------------------------------------------
__global__ void __launch_bounds__(256) final_out(const float* __restrict__ x,
                                                 const _Float16* __restrict__ hg1,
                                                 const float* __restrict__ F_l,
                                                 const float* __restrict__ F_nl,
                                                 const float* __restrict__ F_nlb,
                                                 float* __restrict__ out)
{
    const int b   = blockIdx.x;
    const int tid = threadIdx.x;
    __shared__ float hrow[UNITSC];
    __shared__ float r[2048];

    const int pLast = (T_LEN - 1) & 1;
    hrow[tid] = (float)hg1[(size_t)((pLast * NGROUP + (b >> 5)) * BT + (b & 31)) * UNITSC + tid];
    __syncthreads();

    for (int l = tid; l < 2048; l += 256) {
        float s = 0.f;
#pragma unroll 4
        for (int k = 0; k < UNITSC; ++k) s += hrow[k] * F_nlb[(size_t)k * 2048 + l];
        r[l] = x[(size_t)b * (T_LEN * D_IN) + l] - s;
    }
    __syncthreads();

    for (int d = tid; d < 64; d += 256) {
        float s = 0.f;
#pragma unroll 4
        for (int l = 0; l < 2048; ++l) s += r[l] * F_l[(size_t)l * 64 + d];
        float s2 = 0.f;
#pragma unroll 4
        for (int k = 0; k < UNITSC; ++k) s2 += hrow[k] * F_nl[(size_t)k * 64 + d];
        out[(size_t)b * 64 + d] = s + s2;
    }
}

// ---------------------------------------------------------------------------
extern "C" void kernel_launch(void* const* d_in, const int* in_sizes, int n_in,
                              void* d_out, int out_size, void* d_ws, size_t ws_size,
                              hipStream_t stream)
{
    (void)in_sizes; (void)n_in; (void)out_size; (void)ws_size;

    const float* x      = (const float*)d_in[0];
    const float* sscale = (const float*)d_in[1];
    const float* smin   = (const float*)d_in[2];
    const float* W0     = (const float*)d_in[3];
    const float* U0     = (const float*)d_in[4];
    const float* b0     = (const float*)d_in[5];
    const float* W1     = (const float*)d_in[6];
    const float* U1     = (const float*)d_in[7];
    const float* b1     = (const float*)d_in[8];
    const float* F_l    = (const float*)d_in[9];
    const float* F_nl   = (const float*)d_in[10];
    const float* F_nlb  = (const float*)d_in[11];
    float* out = (float*)d_out;

    char* ws = (char*)d_ws;
    _Float16* wp0 = (_Float16*)(ws + WP0_OFF);
    _Float16* wp1 = (_Float16*)(ws + WP1_OFF);
    _Float16* hg0 = (_Float16*)(ws + HG0_OFF);
    _Float16* hg1 = (_Float16*)(ws + HG1_OFF);
    unsigned* bar = (unsigned*)(ws + BAR_OFF);

    (void)hipFuncSetAttribute(reinterpret_cast<const void*>(&lstm_fused),
                              hipFuncAttributeMaxDynamicSharedMemorySize, SMEM_BYTES);

    hipMemsetAsync(ws + BAR_OFF, 0, BAR_BYTES, stream);

    const int packN = WP0_HALVES + WP1_HALVES;
    pack_weights<<<(packN + 255) / 256, 256, 0, stream>>>(W0, U0, W1, U1, wp0, wp1);
    lstm_fused<<<NGROUP * GROUPW, 256, SMEM_BYTES, stream>>>(x, sscale, smin, b0, b1,
                                                             wp0, wp1, hg0, hg1, bar);
    final_out<<<B_TOT, 256, 0, stream>>>(x, hg1, F_l, F_nl, F_nlb, out);
}
